// ImplicitModel_48009144435571
// MI455X (gfx1250) — matrix operand order repaired and checked by reading, not compiled, not run on hardware
//
#include <hip/hip_runtime.h>
#include <stdint.h>

// Model dims
#define N_DIM 512
#define P_DIM 512
#define Q_DIM 128
#define M_DIM 8192
#define VRAD  0.95f

// GEMM tiling: 256 threads = 8 waves (wave32). Wave grid 4(R) x 2(C),
// wave tile 32x64 => 8 f32 accumulators (2 A-frags x 4 B-frags).
// Block tile: 128 rows x 128 cols, K in chunks of 32 (bf16 WMMA K),
// double-buffered LDS, register-staged global loads, 1 barrier/chunk.
#define BR 128
#define BC 128
#define KC 32
#define SAS 40   // LDS row stride (elements) for A tile, padded vs 64B banks
#define SVS 40   // LDS row stride (elements) for V tile (stored [c][k]), padded

typedef __attribute__((ext_vector_type(8)))  float  v8f;
typedef __attribute__((ext_vector_type(8)))  __bf16 v8bf;
typedef __attribute__((ext_vector_type(16))) __bf16 v16bf;

static __device__ __forceinline__ v16bf cat16(v8bf a, v8bf b) {
  union { v16bf v; v8bf h[2]; } u;
  u.h[0] = a; u.h[1] = b;
  return u.v;
}

#define WMMA_BF16(A, B, Cacc) \
  __builtin_amdgcn_wmma_f32_16x16x32_bf16(false, (A), false, (B), (short)0, (Cacc), false, false)

// ---------------------------------------------------------------------------
// Core: acc += W[tileR:+128, 0:K] * V(k, tileC:+128) with split-bf16
// error compensation (W*V ~= Wh*Vh + Wh*Vl + Wl*Vh, fp32 accumulate).
//   KMAJOR=true : V element (k,c) at V[k*ldv + c]   (X layout)
//   KMAJOR=false: V element (k,c) at V[c*ldv + k]   (U rows = U^T columns)
// Fragment layouts per CDNA5 ISA 7.12.2 (wave32):
//   A 16x32 bf16: lane L (M=L&15) holds K {b..b+7, b+16..b+23}, b=(L>>4)*8
//   B 32x16 bf16: lane L (N=L&15) holds K {b..b+7, b+16..b+23}, b=(L>>4)*8
// ---------------------------------------------------------------------------
template<bool KMAJOR>
static __device__ __forceinline__ void gemm_accumulate(
    const __bf16* __restrict__ Wh, const __bf16* __restrict__ Wl, int ldw,
    const __bf16* __restrict__ Vh, const __bf16* __restrict__ Vl, int ldv,
    int K, int tileR, int tileC,
    __bf16 (*sAh)[BR * SAS], __bf16 (*sAl)[BR * SAS],
    __bf16 (*sVh)[BC * SVS], __bf16 (*sVl)[BC * SVS],
    v8f acc[8])
{
  const int tid  = threadIdx.x;
  const int wave = tid >> 5;
  const int lane = tid & 31;
  const int wr = wave >> 1;     // 0..3 : 32-row slice
  const int wc = wave & 1;      // 0..1 : 64-col slice
  const int ln = lane & 15;
  const int lh = lane >> 4;
  const int akb = lh * 8;
  const int kb2 = lh * 16;

  // staging coordinates
  const int a_row = tid >> 1;            // 0..127
  const int a_ko  = (tid & 1) << 4;      // 0,16
  const int vk_k  = tid >> 3;            // 0..31   (KMAJOR)
  const int vk_co = (tid & 7) << 4;      // 0..112  (KMAJOR)
  const int vc_c  = tid >> 1;            // 0..127  (CMAJOR)
  const int vc_ko = (tid & 1) << 4;      // 0,16    (CMAJOR)

  v8bf rAh0, rAh1, rAl0, rAl1;           // staged W regs
  v8bf rVh0, rVh1, rVl0, rVl1;           // staged V regs

#define LOAD_REGS(KCC)                                                        \
  {                                                                           \
    const size_t ga = (size_t)(tileR + a_row) * ldw + (KCC) + a_ko;           \
    rAh0 = ((const v8bf*)(Wh + ga))[0]; rAh1 = ((const v8bf*)(Wh + ga))[1];   \
    rAl0 = ((const v8bf*)(Wl + ga))[0]; rAl1 = ((const v8bf*)(Wl + ga))[1];   \
    const size_t gv = KMAJOR                                                  \
        ? (size_t)((KCC) + vk_k) * ldv + tileC + vk_co                        \
        : (size_t)(tileC + vc_c) * ldv + (KCC) + vc_ko;                       \
    rVh0 = ((const v8bf*)(Vh + gv))[0]; rVh1 = ((const v8bf*)(Vh + gv))[1];   \
    rVl0 = ((const v8bf*)(Vl + gv))[0]; rVl1 = ((const v8bf*)(Vl + gv))[1];   \
  }

#define STORE_LDS(BUF)                                                        \
  {                                                                           \
    __bf16* dh = sAh[BUF] + a_row * SAS + a_ko;                               \
    __bf16* dl = sAl[BUF] + a_row * SAS + a_ko;                               \
    *(v8bf*)dh = rAh0; *(v8bf*)(dh + 8) = rAh1;                               \
    *(v8bf*)dl = rAl0; *(v8bf*)(dl + 8) = rAl1;                               \
    if (KMAJOR) {                                                             \
      _Pragma("unroll")                                                       \
      for (int j = 0; j < 8; ++j) {                                           \
        sVh[BUF][(vk_co + j)     * SVS + vk_k] = rVh0[j];                     \
        sVh[BUF][(vk_co + 8 + j) * SVS + vk_k] = rVh1[j];                     \
        sVl[BUF][(vk_co + j)     * SVS + vk_k] = rVl0[j];                     \
        sVl[BUF][(vk_co + 8 + j) * SVS + vk_k] = rVl1[j];                     \
      }                                                                       \
    } else {                                                                  \
      __bf16* eh = sVh[BUF] + vc_c * SVS + vc_ko;                             \
      __bf16* el = sVl[BUF] + vc_c * SVS + vc_ko;                             \
      *(v8bf*)eh = rVh0; *(v8bf*)(eh + 8) = rVh1;                             \
      *(v8bf*)el = rVl0; *(v8bf*)(el + 8) = rVl1;                             \
    }                                                                         \
  }

  // prologue: stage chunk 0
  LOAD_REGS(0);
  STORE_LDS(0);
  int buf = 0;

  for (int kc = 0; kc < K; kc += KC) {
    const bool more = (kc + KC) < K;
    if (more) LOAD_REGS(kc + KC);                 // overlap global with compute
    if (kc + 2 * KC < K) {                        // prefetch chunk after next
      __builtin_prefetch(Wh + (size_t)(tileR + a_row) * ldw + kc + 2 * KC + a_ko, 0, 1);
      __builtin_prefetch(Vh + (KMAJOR
          ? (size_t)(kc + 2 * KC + vk_k) * ldv + tileC + vk_co
          : (size_t)(tileC + vc_c) * ldv + kc + 2 * KC + vc_ko), 0, 1);
    }
    __syncthreads();                              // buf's LDS stores visible

    // ---- fragments + WMMA on current buffer ----
    v16bf ah[2], al[2];
#pragma unroll
    for (int s = 0; s < 2; ++s) {
      const __bf16* ph = sAh[buf] + (wr * 32 + s * 16 + ln) * SAS + akb;
      const __bf16* pl = sAl[buf] + (wr * 32 + s * 16 + ln) * SAS + akb;
      ah[s] = cat16(*(const v8bf*)ph, *(const v8bf*)(ph + 16));
      al[s] = cat16(*(const v8bf*)pl, *(const v8bf*)(pl + 16));
    }
#pragma unroll
    for (int j = 0; j < 4; ++j) {
      const __bf16* pb = sVh[buf] + (wc * 64 + j * 16 + ln) * SVS + akb;
      const __bf16* pq = sVl[buf] + (wc * 64 + j * 16 + ln) * SVS + akb;
      v16bf bh = cat16(*(const v8bf*)pb, *(const v8bf*)(pb + 16));
      v16bf bl = cat16(*(const v8bf*)pq, *(const v8bf*)(pq + 16));
#pragma unroll
      for (int s = 0; s < 2; ++s) {
        acc[s * 4 + j] = WMMA_BF16(ah[s], bh, acc[s * 4 + j]);
        acc[s * 4 + j] = WMMA_BF16(ah[s], bl, acc[s * 4 + j]);
        acc[s * 4 + j] = WMMA_BF16(al[s], bh, acc[s * 4 + j]);
      }
    }

    if (more) STORE_LDS(buf ^ 1);   // safe: buf^1 last read before a passed barrier
    buf ^= 1;
  }
#undef LOAD_REGS
#undef STORE_LDS
}

// acc element (s,j,v) -> Out[tileR + wr*32 + s*16 + v + 8*lh,
//                            tileC + wc*64 + j*16 + ln]
#define EPILOGUE_COORDS                                  \
  const int lane = threadIdx.x & 31;                     \
  const int wave = threadIdx.x >> 5;                     \
  const int wr = wave >> 1, wc = wave & 1;               \
  const int ln = lane & 15, lh = lane >> 4;

#define GEMM_SHARED                                            \
  __shared__ __align__(16) __bf16 sAh[2][BR * SAS];            \
  __shared__ __align__(16) __bf16 sAl[2][BR * SAS];            \
  __shared__ __align__(16) __bf16 sVh[2][BC * SVS];            \
  __shared__ __align__(16) __bf16 sVl[2][BC * SVS];

// ---------------------------------------------------------------------------
// BU = B @ U^T   (fp32 out [N, M])
// ---------------------------------------------------------------------------
__global__ __launch_bounds__(256) void k_gemm_bu(
    const uint16_t* __restrict__ Bh_, const uint16_t* __restrict__ Bl_,
    const uint16_t* __restrict__ Uh_, const uint16_t* __restrict__ Ul_,
    float* __restrict__ BU)
{
  GEMM_SHARED
  const __bf16* Bh = (const __bf16*)Bh_; const __bf16* Bl = (const __bf16*)Bl_;
  const __bf16* Uh = (const __bf16*)Uh_; const __bf16* Ul = (const __bf16*)Ul_;
  v8f acc[8] = {};
  const int tileR = blockIdx.y * BR;
  const int tileC = blockIdx.x * BC;
  gemm_accumulate<false>(Bh, Bl, P_DIM, Uh, Ul, P_DIM, P_DIM, tileR, tileC,
                         sAh, sAl, sVh, sVl, acc);
  EPILOGUE_COORDS
#pragma unroll
  for (int s = 0; s < 2; ++s)
#pragma unroll
    for (int j = 0; j < 4; ++j) {
      const int col = tileC + wc * 64 + j * 16 + ln;
#pragma unroll
      for (int v = 0; v < 8; ++v) {
        const int row = tileR + wr * 32 + s * 16 + v + 8 * lh;
        BU[(size_t)row * M_DIM + col] = acc[s * 4 + j][v];
      }
    }
}

// ---------------------------------------------------------------------------
// Y = relu(A @ X + BU), Y stored split as (hi,lo) bf16
// ---------------------------------------------------------------------------
__global__ __launch_bounds__(256) void k_gemm_iter(
    const uint16_t* __restrict__ Ah_, const uint16_t* __restrict__ Al_,
    const uint16_t* __restrict__ Xh_, const uint16_t* __restrict__ Xl_,
    const float* __restrict__ BU,
    uint16_t* __restrict__ Yh_, uint16_t* __restrict__ Yl_)
{
  GEMM_SHARED
  const __bf16* Ah = (const __bf16*)Ah_; const __bf16* Al = (const __bf16*)Al_;
  const __bf16* Xh = (const __bf16*)Xh_; const __bf16* Xl = (const __bf16*)Xl_;
  __bf16* Yh = (__bf16*)Yh_; __bf16* Yl = (__bf16*)Yl_;
  const int tileR = blockIdx.y * BR;
  const int tileC = blockIdx.x * BC;
  EPILOGUE_COORDS
  v8f acc[8];
#pragma unroll
  for (int s = 0; s < 2; ++s)
#pragma unroll
    for (int j = 0; j < 4; ++j) {
      const int col = tileC + wc * 64 + j * 16 + ln;
#pragma unroll
      for (int v = 0; v < 8; ++v) {
        const int row = tileR + wr * 32 + s * 16 + v + 8 * lh;
        acc[s * 4 + j][v] = BU[(size_t)row * M_DIM + col];   // init C with BU
      }
    }
  gemm_accumulate<true>(Ah, Al, N_DIM, Xh, Xl, M_DIM, N_DIM, tileR, tileC,
                        sAh, sAl, sVh, sVl, acc);
#pragma unroll
  for (int s = 0; s < 2; ++s)
#pragma unroll
    for (int j = 0; j < 4; ++j) {
      const int col = tileC + wc * 64 + j * 16 + ln;
#pragma unroll
      for (int v = 0; v < 8; ++v) {
        const int row = tileR + wr * 32 + s * 16 + v + 8 * lh;
        const float x = fmaxf(acc[s * 4 + j][v], 0.0f);
        const __bf16 h = (__bf16)x;
        Yh[(size_t)row * M_DIM + col] = h;
        Yl[(size_t)row * M_DIM + col] = (__bf16)(x - (float)h);
      }
    }
}

// ---------------------------------------------------------------------------
// out = (C @ X + D @ U^T)^T   (fp32 out [M, Q])
// ---------------------------------------------------------------------------
__global__ __launch_bounds__(256) void k_gemm_out(
    const uint16_t* __restrict__ Ch_, const uint16_t* __restrict__ Cl_,
    const uint16_t* __restrict__ Xh_, const uint16_t* __restrict__ Xl_,
    const uint16_t* __restrict__ Dh_, const uint16_t* __restrict__ Dl_,
    const uint16_t* __restrict__ Uh_, const uint16_t* __restrict__ Ul_,
    float* __restrict__ out)
{
  GEMM_SHARED
  const __bf16* Ch = (const __bf16*)Ch_; const __bf16* Cl = (const __bf16*)Cl_;
  const __bf16* Xh = (const __bf16*)Xh_; const __bf16* Xl = (const __bf16*)Xl_;
  const __bf16* Dh = (const __bf16*)Dh_; const __bf16* Dl = (const __bf16*)Dl_;
  const __bf16* Uh = (const __bf16*)Uh_; const __bf16* Ul = (const __bf16*)Ul_;
  v8f acc[8] = {};
  const int tileR = blockIdx.y * BR;
  const int tileC = blockIdx.x * BC;
  gemm_accumulate<true >(Ch, Cl, N_DIM, Xh, Xl, M_DIM, N_DIM, tileR, tileC,
                         sAh, sAl, sVh, sVl, acc);
  gemm_accumulate<false>(Dh, Dl, P_DIM, Uh, Ul, P_DIM, P_DIM, tileR, tileC,
                         sAh, sAl, sVh, sVl, acc);
  EPILOGUE_COORDS
#pragma unroll
  for (int s = 0; s < 2; ++s)
#pragma unroll
    for (int j = 0; j < 4; ++j) {
      const int col = tileC + wc * 64 + j * 16 + ln;      // m
#pragma unroll
      for (int v = 0; v < 8; ++v) {
        const int row = tileR + wr * 32 + s * 16 + v + 8 * lh;   // q
        out[(size_t)col * Q_DIM + row] = acc[s * 4 + j][v];      // transposed
      }
    }
}

// ---------------------------------------------------------------------------
// Row-wise L-inf projection of A (sort + cumsum per row), output split bf16.
// 1 block (256 threads) per row of 512.
// ---------------------------------------------------------------------------
__global__ __launch_bounds__(256) void k_project(
    const float* __restrict__ A, uint16_t* __restrict__ Ah_, uint16_t* __restrict__ Al_)
{
  __shared__ float s[N_DIM];
  __shared__ float cs[N_DIM];
  __shared__ int   rho_sh;
  __shared__ float theta_sh;
  __shared__ int   need_sh;
  __bf16* Ah = (__bf16*)Ah_; __bf16* Al = (__bf16*)Al_;
  const int row = blockIdx.x;
  const int tid = threadIdx.x;
  const float* arow = A + (size_t)row * N_DIM;

  for (int i = tid; i < N_DIM; i += 256) s[i] = fabsf(arow[i]);
  __syncthreads();
  // bitonic sort, descending
  for (int ksz = 2; ksz <= N_DIM; ksz <<= 1) {
    for (int j = ksz >> 1; j > 0; j >>= 1) {
      for (int i = tid; i < N_DIM; i += 256) {
        const int ixj = i ^ j;
        if (ixj > i) {
          const float a = s[i], b = s[ixj];
          const bool desc = (i & ksz) == 0;
          if (desc ? (a < b) : (a > b)) { s[i] = b; s[ixj] = a; }
        }
      }
      __syncthreads();
    }
  }
  // inclusive scan (Hillis-Steele, 512 elems / 256 threads)
  for (int i = tid; i < N_DIM; i += 256) cs[i] = s[i];
  __syncthreads();
  for (int off = 1; off < N_DIM; off <<= 1) {
    const float t0 = (tid >= off) ? cs[tid - off] : 0.0f;
    const float t1 = (tid + 256 >= off) ? cs[tid + 256 - off] : 0.0f;
    __syncthreads();
    cs[tid] += t0;
    cs[tid + 256] += t1;
    __syncthreads();
  }
  if (tid == 0) rho_sh = 0;
  __syncthreads();
  int cnt = 0;
  for (int i = tid; i < N_DIM; i += 256)
    cnt += (s[i] > (cs[i] - VRAD) / (float)(i + 1)) ? 1 : 0;
  atomicAdd(&rho_sh, cnt);
  __syncthreads();
  if (tid == 0) {
    int rho = rho_sh; if (rho < 1) rho = 1;
    theta_sh = (cs[rho - 1] - VRAD) / (float)rho;
    need_sh  = (cs[N_DIM - 1] > VRAD) ? 1 : 0;
  }
  __syncthreads();
  const float theta = theta_sh;
  const int   need  = need_sh;
  for (int i = tid; i < N_DIM; i += 256) {
    const float a = arow[i];
    float r = a;
    if (need) {
      const float m = fmaxf(fabsf(a) - theta, 0.0f);
      r = (a > 0.0f) ? m : ((a < 0.0f) ? -m : 0.0f);
    }
    const __bf16 h = (__bf16)r;
    Ah[(size_t)row * N_DIM + i] = h;
    Al[(size_t)row * N_DIM + i] = (__bf16)(r - (float)h);
  }
}

// fp32 -> (hi, lo) bf16 split
__global__ __launch_bounds__(256) void k_split(
    const float* __restrict__ in, uint16_t* __restrict__ hi_, uint16_t* __restrict__ lo_, int n)
{
  __bf16* hi = (__bf16*)hi_; __bf16* lo = (__bf16*)lo_;
  const int i = blockIdx.x * 256 + threadIdx.x;
  if (i < n) {
    const float x = in[i];
    const __bf16 h = (__bf16)x;
    hi[i] = h;
    lo[i] = (__bf16)(x - (float)h);
  }
}

// X1 = relu(BU) split  (first fixed-point evaluation, X0 == 0)
__global__ __launch_bounds__(256) void k_relu_split(
    const float* __restrict__ in, uint16_t* __restrict__ hi_, uint16_t* __restrict__ lo_, int n)
{
  __bf16* hi = (__bf16*)hi_; __bf16* lo = (__bf16*)lo_;
  const int i = blockIdx.x * 256 + threadIdx.x;
  if (i < n) {
    const float x = fmaxf(in[i], 0.0f);
    const __bf16 h = (__bf16)x;
    hi[i] = h;
    lo[i] = (__bf16)(x - (float)h);
  }
}

// ---------------------------------------------------------------------------
extern "C" void kernel_launch(void* const* d_in, const int* in_sizes, int n_in,
                              void* d_out, int out_size, void* d_ws, size_t ws_size,
                              hipStream_t stream)
{
  (void)in_sizes; (void)n_in; (void)out_size; (void)ws_size;
  const float* U = (const float*)d_in[0];   // [M, P]
  const float* A = (const float*)d_in[1];   // [N, N]
  const float* B = (const float*)d_in[2];   // [N, P]
  const float* C = (const float*)d_in[3];   // [Q, N]
  const float* D = (const float*)d_in[4];   // [Q, P]
  float* out = (float*)d_out;               // [M, Q]

  uint8_t* w = (uint8_t*)d_ws;
  auto take = [&](size_t bytes) -> void* {
    void* p = (void*)w;
    w += (bytes + 255) & ~(size_t)255;
    return p;
  };
  uint16_t* Ahp = (uint16_t*)take((size_t)N_DIM * N_DIM * 2);
  uint16_t* Alp = (uint16_t*)take((size_t)N_DIM * N_DIM * 2);
  uint16_t* Bhp = (uint16_t*)take((size_t)N_DIM * P_DIM * 2);
  uint16_t* Blp = (uint16_t*)take((size_t)N_DIM * P_DIM * 2);
  uint16_t* Chp = (uint16_t*)take((size_t)Q_DIM * N_DIM * 2);
  uint16_t* Clp = (uint16_t*)take((size_t)Q_DIM * N_DIM * 2);
  uint16_t* Dhp = (uint16_t*)take((size_t)Q_DIM * P_DIM * 2);
  uint16_t* Dlp = (uint16_t*)take((size_t)Q_DIM * P_DIM * 2);
  uint16_t* Uhp = (uint16_t*)take((size_t)M_DIM * P_DIM * 2);
  uint16_t* Ulp = (uint16_t*)take((size_t)M_DIM * P_DIM * 2);
  float*    BUp = (float*)   take((size_t)N_DIM * M_DIM * 4);
  uint16_t* X0h = (uint16_t*)take((size_t)N_DIM * M_DIM * 2);
  uint16_t* X0l = (uint16_t*)take((size_t)N_DIM * M_DIM * 2);
  uint16_t* X1h = (uint16_t*)take((size_t)N_DIM * M_DIM * 2);
  uint16_t* X1l = (uint16_t*)take((size_t)N_DIM * M_DIM * 2);

  // 1) project A rows onto L1 ball (=> ||A||inf <= 0.95), split bf16
  k_project<<<N_DIM, 256, 0, stream>>>(A, Ahp, Alp);
  // 2) split B, C, D, U
  k_split<<<(N_DIM * P_DIM) / 256, 256, 0, stream>>>(B, Bhp, Blp, N_DIM * P_DIM);
  k_split<<<(Q_DIM * N_DIM) / 256, 256, 0, stream>>>(C, Chp, Clp, Q_DIM * N_DIM);
  k_split<<<(Q_DIM * P_DIM) / 256, 256, 0, stream>>>(D, Dhp, Dlp, Q_DIM * P_DIM);
  k_split<<<(M_DIM * P_DIM) / 256, 256, 0, stream>>>(U, Uhp, Ulp, M_DIM * P_DIM);
  // 3) BU = B @ U^T
  dim3 gNM(M_DIM / BC, N_DIM / BR);
  k_gemm_bu<<<gNM, 256, 0, stream>>>(Bhp, Blp, Uhp, Ulp, BUp);
  // 4) eval 1: X = relu(BU)
  k_relu_split<<<(N_DIM * M_DIM) / 256, 256, 0, stream>>>(BUp, X0h, X0l, N_DIM * M_DIM);
  // 5) evals 2..50: X <- relu(A @ X + BU)   (contraction never hits 3e-6 in 50 iters)
  uint16_t* curh = X0h; uint16_t* curl = X0l;
  uint16_t* nxth = X1h; uint16_t* nxtl = X1l;
  for (int it = 0; it < 49; ++it) {
    k_gemm_iter<<<gNM, 256, 0, stream>>>(Ahp, Alp, curh, curl, BUp, nxth, nxtl);
    uint16_t* th = curh; uint16_t* tl = curl;
    curh = nxth; curl = nxtl;
    nxth = th;   nxtl = tl;
  }
  // 6) out = (C @ X + D @ U^T)^T
  dim3 gQM(M_DIM / BC, Q_DIM / BR);
  k_gemm_out<<<gQM, 256, 0, stream>>>(Chp, Clp, curh, curl, Dhp, Dlp, Uhp, Ulp, out);
}
